// ManifoldProjection_49289044689192
// MI455X (gfx1250) — matrix operand order repaired and checked by reading, not compiled, not run on hardware
//
#include <hip/hip_runtime.h>
#include <cstdint>

// ManifoldProjection for MI455X (gfx1250):
//   1) Sinkhorn-Knopp (fp32, L2-resident, fused row/col normalize passes)
//   2) fp32->bf16 conversion of ds_matrix and residual
//   3) bf16 WMMA GEMM (16x16x32), block tile 128x256, wave tile 64x64,
//      TDM double-buffered LDS tiles; epilogue applies 0.01*alpha.

#define EPS 1e-8f
#define DIM 4096
#define MROWS 8192          // 4 * 2048
#define SK_ITERS 20

typedef float  v8f   __attribute__((ext_vector_type(8)));
typedef __bf16 v4bf  __attribute__((ext_vector_type(4)));
typedef __bf16 v8bf  __attribute__((ext_vector_type(8)));
typedef __bf16 v16bf __attribute__((ext_vector_type(16)));
typedef unsigned int u32x4 __attribute__((ext_vector_type(4)));
typedef int i32x4 __attribute__((ext_vector_type(4)));
typedef int i32x8 __attribute__((ext_vector_type(8)));

// ---------------------------------------------------------------------------
// Phase 1a: x_pos = exp(x - rowmax); rsinv[r] = 1/(rowsum + eps)
// ---------------------------------------------------------------------------
__global__ __launch_bounds__(256) void k_exp_rowsum(const float* __restrict__ W,
                                                    float* __restrict__ M,
                                                    float* __restrict__ rsinv) {
  const int r = blockIdx.x;
  const int t = threadIdx.x;
  const float* row = W + (size_t)r * DIM;
  __shared__ float red[256];

  float mx = -3.4e38f;
  for (int j = t; j < DIM; j += 256) mx = fmaxf(mx, row[j]);
  red[t] = mx;
  __syncthreads();
  for (int s = 128; s > 0; s >>= 1) {
    if (t < s) red[t] = fmaxf(red[t], red[t + s]);
    __syncthreads();
  }
  mx = red[0];
  __syncthreads();

  float sum = 0.f;
  float* out = M + (size_t)r * DIM;
  for (int j = t; j < DIM; j += 256) {
    float e = __expf(row[j] - mx);
    out[j] = e;
    sum += e;
  }
  red[t] = sum;
  __syncthreads();
  for (int s = 128; s > 0; s >>= 1) {
    if (t < s) red[t] += red[t + s];
    __syncthreads();
  }
  if (t == 0) rsinv[r] = 1.0f / (red[0] + EPS);
}

// ---------------------------------------------------------------------------
// Phase 1b: pass A — apply row scale, produce csinv (block owns 32 whole cols)
// ---------------------------------------------------------------------------
__global__ __launch_bounds__(256) void k_passA(float* __restrict__ M,
                                               const float* __restrict__ rsinv,
                                               float* __restrict__ csinv) {
  const int tc = threadIdx.x & 31;
  const int tr = threadIdx.x >> 5;          // 0..7
  const int c  = blockIdx.x * 32 + tc;
  float part = 0.f;
  for (int r = tr; r < DIM; r += 8) {
    size_t idx = (size_t)r * DIM + c;
    float v = M[idx] * rsinv[r];
    M[idx] = v;
    part += v;
  }
  __shared__ float red[8][32];
  red[tr][tc] = part;
  __syncthreads();
  if (tr == 0) {
    float s = 0.f;
#pragma unroll
    for (int i = 0; i < 8; ++i) s += red[i][tc];
    csinv[c] = 1.0f / (s + EPS);
  }
}

// ---------------------------------------------------------------------------
// Phase 1c: pass B — apply col scale, produce rsinv (block owns 32 whole rows)
// ---------------------------------------------------------------------------
__global__ __launch_bounds__(256) void k_passB(float* __restrict__ M,
                                               const float* __restrict__ csinv,
                                               float* __restrict__ rsinv) {
  const int r0 = blockIdx.x * 32;
  const int t = threadIdx.x;
  __shared__ float red[256];
  const float4* cs4 = (const float4*)csinv;
  for (int rr = 0; rr < 32; ++rr) {
    const int r = r0 + rr;
    float4* rowp = (float4*)(M + (size_t)r * DIM);
    float part = 0.f;
    for (int j = t; j < DIM / 4; j += 256) {
      float4 v = rowp[j];
      float4 c = cs4[j];
      v.x *= c.x; v.y *= c.y; v.z *= c.z; v.w *= c.w;
      rowp[j] = v;
      part += v.x + v.y + v.z + v.w;
    }
    red[t] = part;
    __syncthreads();
    for (int s = 128; s > 0; s >>= 1) {
      if (t < s) red[t] += red[t + s];
      __syncthreads();
    }
    if (t == 0) rsinv[r] = 1.0f / (red[0] + EPS);
    __syncthreads();
  }
}

// ---------------------------------------------------------------------------
// Phase 2: fp32 -> bf16 (vec4)
// ---------------------------------------------------------------------------
__global__ __launch_bounds__(256) void k_cvt_bf16(const float* __restrict__ src,
                                                  __bf16* __restrict__ dst,
                                                  long long n4) {
  long long i = (long long)blockIdx.x * 256 + threadIdx.x;
  if (i >= n4) return;
  float4 v = ((const float4*)src)[i];
  v4bf o = { (__bf16)v.x, (__bf16)v.y, (__bf16)v.z, (__bf16)v.w };
  ((v4bf*)dst)[i] = o;
}

// ---------------------------------------------------------------------------
// TDM: load a (tile_rows x 32) bf16 tile (memory row stride DIM elems) into
// LDS with 16B pad per 64B row => 80B LDS row stride.
// ---------------------------------------------------------------------------
__device__ __forceinline__ void tdm_load_tile(const __bf16* base, unsigned row0,
                                              unsigned nrows, unsigned k0,
                                              unsigned lds_addr,
                                              unsigned tile_rows) {
  unsigned long long ga = (unsigned long long)(uintptr_t)base +
      (((unsigned long long)row0 * (unsigned long long)DIM + k0) * 2ull);
  u32x4 g0;
  g0[0] = 1u;                                            // count=1, user D#
  g0[1] = lds_addr;                                      // LDS byte address
  g0[2] = (unsigned)ga;                                  // global addr lo32
  g0[3] = (unsigned)((ga >> 32) & 0x1FFFFFFull) | (2u << 30);  // hi25 | type=2
  i32x8 g1;
  // data_size=2B(1<<16), pad_enable(1<<20), pad_interval=16dw(3<<22),
  // pad_amount=4dw(3<<25); workgroup_mask=0
  g1[0] = (int)((1u << 16) | (1u << 20) | (3u << 22) | (3u << 25));
  g1[1] = (int)((unsigned)DIM << 16);                    // tensor_dim0 lo16
  g1[2] = (int)(((unsigned)DIM >> 16) | (nrows << 16));  // td0 hi16 | td1 lo16
  g1[3] = (int)((nrows >> 16) | (32u << 16));            // td1 hi16 | tile_dim0=32
  g1[4] = (int)tile_rows;                                // tile_dim1, tile_dim2=0
  g1[5] = DIM;                                           // tensor_dim0_stride lo32
  g1[6] = 0;
  g1[7] = 0;
  i32x4 z4 = { 0, 0, 0, 0 };
#if defined(__clang_major__) && (__clang_major__ >= 23)
  i32x8 z8 = { 0, 0, 0, 0, 0, 0, 0, 0 };
  __builtin_amdgcn_tensor_load_to_lds(g0, g1, z4, z4, z8, 0);
#else
  __builtin_amdgcn_tensor_load_to_lds(g0, g1, z4, z4, 0);
#endif
}

// ---------------------------------------------------------------------------
// Phase 3: C[M,N] = (0.01*alpha) * A[M,K] * W[N,K]^T  (bf16 in, fp32 out)
// block tile 128(M) x 256(N), BK=32; 8 waves (2x4), wave tile 64x64 = 4x4 accs
// (16 WMMA per K-step per wave). TDM double-buffered LDS (61,440 B total).
// ---------------------------------------------------------------------------
#define LDS_ROW 80u          // 32*2 bytes + 16B pad
#define A_BYTES 10240u       // 128 * 80
#define B_BYTES 20480u       // 256 * 80
#define BUF_BYTES 30720u     // A_BYTES + B_BYTES

__global__ __launch_bounds__(256) void k_gemm_wmma(const __bf16* __restrict__ A,
                                                   const __bf16* __restrict__ Wb,
                                                   const float* __restrict__ alpha,
                                                   float* __restrict__ C) {
  __shared__ __align__(256) unsigned char smem[2 * BUF_BYTES];
  const int tid  = threadIdx.x;
  const int wave = tid >> 5;
  const int lane = tid & 31;
  const int wm = wave >> 2;                 // 0..1 (M half)
  const int wn = wave & 3;                  // 0..3 (N quarter)
  const unsigned bm = blockIdx.y * 128u;
  const unsigned bn = blockIdx.x * 256u;

  const v8f vzero = { 0.f, 0.f, 0.f, 0.f, 0.f, 0.f, 0.f, 0.f };
  v8f acc[4][4];
#pragma unroll
  for (int t = 0; t < 4; ++t)
#pragma unroll
    for (int u = 0; u < 4; ++u) acc[t][u] = vzero;

  const unsigned ldsBase = (unsigned)(uintptr_t)&smem[0];

  const int KIT = DIM / 32;                 // 128 K-steps
  if (wave == 0) {
    tdm_load_tile(A,  bm, MROWS, 0u, ldsBase,            128u);
    tdm_load_tile(Wb, bn, DIM,   0u, ldsBase + A_BYTES,  256u);
    __builtin_amdgcn_s_wait_tensorcnt(0);
  }
  __syncthreads();

  const int mlane = lane & 15;
  const int lh = lane >> 4;                 // K-half select

  for (int kk = 0; kk < KIT; ++kk) {
    const unsigned cur = (unsigned)(kk & 1) * BUF_BYTES;
    const unsigned nxt = cur ^ BUF_BYTES;
    if (wave == 0 && (kk + 1) < KIT) {
      tdm_load_tile(A,  bm, MROWS, (unsigned)(kk + 1) * 32u, ldsBase + nxt, 128u);
      tdm_load_tile(Wb, bn, DIM,   (unsigned)(kk + 1) * 32u,
                    ldsBase + nxt + A_BYTES, 256u);
    }

    const unsigned char* tA = smem + cur;
    const unsigned char* tB = smem + cur + A_BYTES;

    v16bf a[4], b[4];
    // A 16x32 bf16 frag: lane<16 -> M=lane, K0..7 & K16..23; lane>=16 -> K8..15 & K24..31
#pragma unroll
    for (int t = 0; t < 4; ++t) {
      const unsigned char* p =
          tA + (unsigned)(wm * 64 + t * 16 + mlane) * LDS_ROW + (unsigned)lh * 16u;
      union { v16bf v; v8bf h[2]; } f;
      f.h[0] = *(const v8bf*)p;
      f.h[1] = *(const v8bf*)(p + 32);
      a[t] = f.v;
    }
    // B 32x16 bf16 frag: lane<16 -> N=lane, K0..7 & K16..23; lane>=16 -> K8..15 & K24..31
#pragma unroll
    for (int u = 0; u < 4; ++u) {
      const unsigned char* p =
          tB + (unsigned)(wn * 64 + u * 16 + mlane) * LDS_ROW + (unsigned)lh * 16u;
      union { v16bf v; v8bf h[2]; } f;
      f.h[0] = *(const v8bf*)p;
      f.h[1] = *(const v8bf*)(p + 32);
      b[u] = f.v;
    }

#pragma unroll
    for (int u = 0; u < 4; ++u)
#pragma unroll
      for (int t = 0; t < 4; ++t)
        acc[t][u] = __builtin_amdgcn_wmma_f32_16x16x32_bf16(
            false, a[t], false, b[u], (short)0, acc[t][u], false, false);

    if (wave == 0 && (kk + 1) < KIT) __builtin_amdgcn_s_wait_tensorcnt(0);
    __syncthreads();
  }

  const float scale = 0.01f * alpha[0];
  // C/D layout: VGPR r -> lanes 0-15: (M=r, N=lane); lanes 16-31: (M=8+r, N=lane-16)
#pragma unroll
  for (int t = 0; t < 4; ++t)
#pragma unroll
    for (int u = 0; u < 4; ++u) {
      const unsigned col = bn + (unsigned)(wn * 64 + u * 16 + mlane);
#pragma unroll
      for (int r = 0; r < 8; ++r) {
        const unsigned row = bm + (unsigned)(wm * 64 + t * 16 + lh * 8 + r);
        C[(size_t)row * DIM + col] = scale * acc[t][u][r];
      }
    }
}

// ---------------------------------------------------------------------------
extern "C" void kernel_launch(void* const* d_in, const int* in_sizes, int n_in,
                              void* d_out, int out_size, void* d_ws, size_t ws_size,
                              hipStream_t stream) {
  const float* residual = (const float*)d_in[0];  // [4,2048,4096] fp32
  const float* hres     = (const float*)d_in[1];  // [4096,4096] fp32
  const float* alpha    = (const float*)d_in[2];  // [1] fp32
  float* out = (float*)d_out;                     // [4,2048,4096] fp32

  char* ws = (char*)d_ws;
  float* Mws   = (float*)ws;  ws += (size_t)DIM * DIM * 4;   // 64 MB fp32 working matrix
  float* rsinv = (float*)ws;  ws += (size_t)DIM * 4;
  float* csinv = (float*)ws;  ws += (size_t)DIM * 4;
  ws = (char*)(((uintptr_t)ws + 255) & ~(uintptr_t)255);
  __bf16* Wb = (__bf16*)ws;   ws += (size_t)DIM * DIM * 2;   // 32 MB bf16 ds_matrix
  __bf16* Ab = (__bf16*)ws;                                  // 64 MB bf16 residual

  // Sinkhorn-Knopp (fused passes, L2-resident)
  k_exp_rowsum<<<DIM, 256, 0, stream>>>(hres, Mws, rsinv);
  for (int it = 0; it < SK_ITERS; ++it) {
    k_passA<<<DIM / 32, 256, 0, stream>>>(Mws, rsinv, csinv);
    k_passB<<<DIM / 32, 256, 0, stream>>>(Mws, csinv, rsinv);
  }

  // bf16 conversions for the WMMA GEMM
  long long nW4 = (long long)DIM * DIM / 4;
  long long nA4 = (long long)MROWS * DIM / 4;
  k_cvt_bf16<<<(unsigned)((nW4 + 255) / 256), 256, 0, stream>>>(Mws, Wb, nW4);
  k_cvt_bf16<<<(unsigned)((nA4 + 255) / 256), 256, 0, stream>>>(residual, Ab, nA4);

  // GEMM: (8192x4096) x (4096x4096)^T, scaled by 0.01*alpha
  dim3 grid(DIM / 256, MROWS / 128);
  k_gemm_wmma<<<grid, 256, 0, stream>>>(Ab, Wb, alpha, out);
}
